// EquiConv_17394617549524
// MI455X (gfx1250) — matrix operand order repaired and checked by reading, not compiled, not run on hardware
//
#include <hip/hip_runtime.h>
#include <hip/hip_bf16.h>

// ---------------- problem constants ----------------
#define MUL_S  128
#define MUL_V  64
#define FC_IN  128
#define FC_HID 64
#define LEN_W  192
#define ROW_IN1 320   // fea_in1 row stride (floats)
#define ROW_OUT 320   // output row stride (floats)

#define INV_S 0.08838834764831845f   // 1/sqrt(128)
#define INV_V 0.125f                 // 1/sqrt(64)
#define SQ2   0.7071067811865476f    // 1/sqrt(2)
#define SQ3   0.5773502691896258f    // 1/sqrt(3)

#define NW 8            // waves per block
#define NTHR (NW * 32)

typedef __attribute__((ext_vector_type(16))) _Float16 v16h;
typedef __attribute__((ext_vector_type(8)))  float    v8f;
typedef __attribute__((ext_vector_type(4)))  float    v4f;
typedef unsigned int u32x4 __attribute__((ext_vector_type(4)));
typedef int          i32x8 __attribute__((ext_vector_type(8)));
typedef int          i32x4 __attribute__((ext_vector_type(4)));

union HFrag { v16h v; uint4 q[2]; };

// ---------------- LDS layout (element offsets) ----------------
constexpr int OFF_WT0 = 0;                      // w1_p0^T [128][128]
constexpr int OFF_WT1 = OFF_WT0 + 128 * 128;    // w1_p1^T [64][128]
constexpr int OFF_WT2 = OFF_WT1 + 64 * 128;     // w1_p2^T [64][128]
constexpr int OFF_WT3 = OFF_WT2 + 64 * 128;     // w1_p3^T [64][64]
constexpr int OFF_WT4 = OFF_WT3 + 64 * 64;      // w1_p4^T [128][64]
constexpr int OFF_WT5 = OFF_WT4 + 128 * 64;     // w1_p5^T [64][64]
constexpr int OFF_WF0 = OFF_WT5 + 64 * 64;      // fc_w0^T [64][128]
constexpr int OFF_WF1 = OFF_WF0 + 64 * 128;     // fc_w1^T [64][64]
constexpr int OFF_WF2 = OFF_WF1 + 64 * 64;      // fc_w2^T [192][64]
constexpr int NHALF_W = OFF_WF2 + 192 * 64;     // 73728 halves = 147456 B

constexpr int FSEC_BYTES_OFF = NHALF_W * 2;     // float section (16B aligned)
constexpr int NFLOAT_SMALL = 832;               // w2_* + biases
constexpr int BLOB_BYTES = FSEC_BYTES_OFF + NFLOAT_SMALL * 4;  // 150784 (mult of 8 and 16)

constexpr int WAVE_BYTES_OFF = BLOB_BYTES;
// per-wave staging: sx1v[3][16][64] f16 + sdv[16][64] f16 + sh[16][64] f16 + sx2[16][4] f32
constexpr int WAVE_BYTES = 5120 * 2 + 64 * 4;   // 10496 B (mult of 16)
constexpr size_t SMEM_BYTES = (size_t)WAVE_BYTES_OFF + (size_t)NW * WAVE_BYTES; // ~229.3 KB

// ---------------- helpers ----------------
__device__ __forceinline__ v8f zf() {
  v8f r;
#pragma unroll
  for (int i = 0; i < 8; ++i) r[i] = 0.f;
  return r;
}

__device__ __forceinline__ float sigmoidf_(float x) { return 1.0f / (1.0f + __expf(-x)); }

__device__ __forceinline__ v8f wmma_f16(v16h a, v16h b, v8f c) {
  // D = A(16x32,f16) x B(32x16,f16) + C(16x16,f32)
  return __builtin_amdgcn_wmma_f32_16x16x32_f16(false, a, false, b, (short)0, c, false, false);
}

// A-fragment from row-major f16 array [rows][ld]; elements 0..7 = [k0..k0+7], 8..15 = [k0+16..k0+23]
__device__ __forceinline__ v16h fragA_lds(const _Float16* base, int ld, int row, int k0) {
  HFrag f;
  const uint4* p = reinterpret_cast<const uint4*>(base + row * ld + k0);
  f.q[0] = p[0];
  f.q[1] = p[2];        // +16 halves = +32 bytes
  return f.v;
}

// B-fragment from transposed weight wT[N][ldk]; elements 0..7 = [k0..k0+7], 8..15 = [k0+16..k0+23], k0 = kb - (kb & 16) / 2
__device__ __forceinline__ v16h fragB_lds(const _Float16* wT, int ldk, int n, int kb) {
  HFrag f;
  const uint4* p = reinterpret_cast<const uint4*>(wT + n * ldk + kb - ((kb & 16) >> 1));
  f.q[0] = p[0];
  f.q[1] = p[2];
  return f.v;
}

// A-fragment gathered from global f32 row-major [*, ld] with fp32->fp16 convert
__device__ __forceinline__ v16h fragA_gbl(const float* __restrict__ src, long row, int ld,
                                          int k0, bool valid) {
  v16h f;
  if (valid) {
    const float* p = src + row * (long)ld + k0;
    v4f a = *reinterpret_cast<const v4f*>(p);
    v4f b = *reinterpret_cast<const v4f*>(p + 4);
    v4f c = *reinterpret_cast<const v4f*>(p + 16);
    v4f d = *reinterpret_cast<const v4f*>(p + 20);
#pragma unroll
    for (int i = 0; i < 4; ++i) {
      f[i]      = (_Float16)a[i];
      f[4 + i]  = (_Float16)b[i];
      f[8 + i]  = (_Float16)c[i];
      f[12 + i] = (_Float16)d[i];
    }
  } else {
#pragma unroll
    for (int i = 0; i < 16; ++i) f[i] = (_Float16)0.f;
  }
  return f;
}

// transpose-convert weight (K,N) fp32 row-major -> wT[N][K] f16 (K power of 2 -> shifts)
template <int K, int N, typename DST>
__device__ __forceinline__ void cvtWT(const float* __restrict__ g, DST* dst, int t, int stride) {
  constexpr int sz = K * N;
  for (int idx = t; idx < sz; idx += stride) {
    int n = idx / K;
    int k = idx & (K - 1);
    dst[idx] = (DST)g[k * N + n];
  }
}

__device__ __forceinline__ void copyF(const float* __restrict__ g, float* dst, int n,
                                      int t, int stride) {
  for (int i = t; i < n; i += stride) dst[i] = g[i];
}

// shared by the one-shot blob builder and the in-kernel fallback path
template <typename HT>
__device__ __forceinline__ void buildWeightSections(
    HT* hbase, float* fsec, int t, int stride,
    const float* w1_p0, const float* w2_p0, const float* w1_p1, const float* w2_p1,
    const float* w1_p2, const float* w2_p2, const float* w1_p3, const float* w2_p3,
    const float* w1_p4, const float* w2_p4, const float* w1_p5, const float* w2_p5,
    const float* fc_w0, const float* fc_b0, const float* fc_w1, const float* fc_b1,
    const float* fc_w2, const float* fc_b2) {
  cvtWT<128, 128>(w1_p0, hbase + OFF_WT0, t, stride);
  cvtWT<128, 64>(w1_p1, hbase + OFF_WT1, t, stride);
  cvtWT<128, 64>(w1_p2, hbase + OFF_WT2, t, stride);
  cvtWT<64, 64>(w1_p3, hbase + OFF_WT3, t, stride);
  cvtWT<64, 128>(w1_p4, hbase + OFF_WT4, t, stride);
  cvtWT<64, 64>(w1_p5, hbase + OFF_WT5, t, stride);
  cvtWT<128, 64>(fc_w0, hbase + OFF_WF0, t, stride);
  cvtWT<64, 64>(fc_w1, hbase + OFF_WF1, t, stride);
  cvtWT<64, 192>(fc_w2, hbase + OFF_WF2, t, stride);
  copyF(w2_p0, fsec + 0, 128, t, stride);
  copyF(w2_p1, fsec + 128, 64, t, stride);
  copyF(w2_p2, fsec + 192, 64, t, stride);
  copyF(w2_p3, fsec + 256, 64, t, stride);
  copyF(w2_p4, fsec + 320, 128, t, stride);
  copyF(w2_p5, fsec + 448, 64, t, stride);
  copyF(fc_b0, fsec + 512, 64, t, stride);
  copyF(fc_b1, fsec + 576, 64, t, stride);
  copyF(fc_b2, fsec + 640, 192, t, stride);
}

// ---------------- one-shot kernel: build f16 weight blob in workspace ----------------
__global__ void build_blob_kernel(
    const float* __restrict__ w1_p0, const float* __restrict__ w2_p0,
    const float* __restrict__ w1_p1, const float* __restrict__ w2_p1,
    const float* __restrict__ w1_p2, const float* __restrict__ w2_p2,
    const float* __restrict__ w1_p3, const float* __restrict__ w2_p3,
    const float* __restrict__ w1_p4, const float* __restrict__ w2_p4,
    const float* __restrict__ w1_p5, const float* __restrict__ w2_p5,
    const float* __restrict__ fc_w0, const float* __restrict__ fc_b0,
    const float* __restrict__ fc_w1, const float* __restrict__ fc_b1,
    const float* __restrict__ fc_w2, const float* __restrict__ fc_b2,
    char* __restrict__ blob) {
  int t = blockIdx.x * blockDim.x + threadIdx.x;
  int stride = gridDim.x * blockDim.x;
  buildWeightSections(reinterpret_cast<_Float16*>(blob),
                      reinterpret_cast<float*>(blob + FSEC_BYTES_OFF), t, stride,
                      w1_p0, w2_p0, w1_p1, w2_p1, w1_p2, w2_p2, w1_p3, w2_p3,
                      w1_p4, w2_p4, w1_p5, w2_p5, fc_w0, fc_b0, fc_w1, fc_b1, fc_w2, fc_b2);
}

// ---------------- fused kernel ----------------
__launch_bounds__(NTHR, 1)
__global__ void equiconv_wmma_kernel(
    const float* __restrict__ fea_in1, const float* __restrict__ fea_in2,
    const float* __restrict__ fea_weight,
    const float* __restrict__ w1_p0, const float* __restrict__ w2_p0,
    const float* __restrict__ w1_p1, const float* __restrict__ w2_p1,
    const float* __restrict__ w1_p2, const float* __restrict__ w2_p2,
    const float* __restrict__ w1_p3, const float* __restrict__ w2_p3,
    const float* __restrict__ w1_p4, const float* __restrict__ w2_p4,
    const float* __restrict__ w1_p5, const float* __restrict__ w2_p5,
    const float* __restrict__ fc_w0, const float* __restrict__ fc_b0,
    const float* __restrict__ fc_w1, const float* __restrict__ fc_b1,
    const float* __restrict__ fc_w2, const float* __restrict__ fc_b2,
    const char* __restrict__ wblob, int mode,
    float* __restrict__ out, long Etot)
{
  extern __shared__ __align__(16) char smem[];
  _Float16* hbase = reinterpret_cast<_Float16*>(smem);
  _Float16* wT0 = hbase + OFF_WT0;
  _Float16* wT1 = hbase + OFF_WT1;
  _Float16* wT2 = hbase + OFF_WT2;
  _Float16* wT3 = hbase + OFF_WT3;
  _Float16* wT4 = hbase + OFF_WT4;
  _Float16* wT5 = hbase + OFF_WT5;
  _Float16* wF0 = hbase + OFF_WF0;
  _Float16* wF1 = hbase + OFF_WF1;
  _Float16* wF2 = hbase + OFF_WF2;
  float* fsec  = reinterpret_cast<float*>(smem + FSEC_BYTES_OFF);
  float* sw2_0 = fsec + 0;    // 128
  float* sw2_1 = fsec + 128;  // 64
  float* sw2_2 = fsec + 192;  // 64
  float* sw2_3 = fsec + 256;  // 64
  float* sw2_4 = fsec + 320;  // 128
  float* sw2_5 = fsec + 448;  // 64
  float* sb0   = fsec + 512;  // 64
  float* sb1   = fsec + 576;  // 64
  float* sb2   = fsec + 640;  // 192

  const int tid = threadIdx.x;
  const int wave = tid >> 5;

  // ---- weight preload into LDS ----
  if (mode == 1) {
#if __has_builtin(__builtin_amdgcn_tensor_load_to_lds) && __has_builtin(__builtin_amdgcn_s_wait_tensorcnt)
    // Tensor Data Mover: one wave DMA's the whole pre-built 150784B blob into LDS.
    if (wave == 0) {
      unsigned ldsBase = (unsigned)__builtin_amdgcn_groupstaticsize();  // dynamic LDS start
      unsigned long long ga = (unsigned long long)wblob;
      constexpr unsigned NE = (unsigned)(BLOB_BYTES / 8);  // 18848 8-byte elements
      u32x4 g0;
      g0[0] = 1u;                                   // count=1 (valid user descriptor)
      g0[1] = ldsBase;                              // lds_addr (bytes)
      g0[2] = (unsigned)(ga & 0xffffffffull);       // global_addr[31:0]
      g0[3] = (unsigned)((ga >> 32) & 0x01ffffffull) | (2u << 30);  // addr[56:32] | type=2
      i32x8 g1;
      g1[0] = (int)(3u << 16);                      // wg_mask=0, data_size=8B
      g1[1] = (int)((NE & 0xffffu) << 16);          // tensor_dim0[15:0]
      g1[2] = (int)(((NE >> 16) & 0xffffu) | (1u << 16));   // dim0[31:16] | tensor_dim1=1 lo
      g1[3] = (int)((NE & 0xffffu) << 16);          // dim1 hi=0 | tile_dim0=NE
      g1[4] = 1;                                    // tile_dim1=1, tile_dim2=0
      g1[5] = (int)NE;                              // tensor_dim0_stride[31:0]
      g1[6] = 0;                                    // stride0 hi | stride1 lo
      g1[7] = 0;
      i32x4 g2; g2[0] = 0; g2[1] = 0; g2[2] = 0; g2[3] = 0;   // 2D: groups 2/3 unused
      i32x4 g3; g3[0] = 0; g3[1] = 0; g3[2] = 0; g3[3] = 0;
      i32x8 g4;                                      // extra group of 6-arg builtin: zero-filled
#pragma unroll
      for (int i = 0; i < 8; ++i) g4[i] = 0;
      __builtin_amdgcn_tensor_load_to_lds(g0, g1, g2, g3, g4, 0);
      __builtin_amdgcn_s_wait_tensorcnt(0);
    }
#else
    // fallback: contiguous coalesced copy of the pre-built blob
    {
      const uint4* src = reinterpret_cast<const uint4*>(wblob);
      uint4* dst = reinterpret_cast<uint4*>(smem);
      for (int i = tid; i < BLOB_BYTES / 16; i += NTHR) dst[i] = src[i];
    }
#endif
  } else {
    // no workspace: transpose+convert straight from the fp32 weights
    buildWeightSections(hbase, fsec, tid, NTHR,
                        w1_p0, w2_p0, w1_p1, w2_p1, w1_p2, w2_p2, w1_p3, w2_p3,
                        w1_p4, w2_p4, w1_p5, w2_p5, fc_w0, fc_b0, fc_w1, fc_b1, fc_w2, fc_b2);
  }
  __syncthreads();

  const int lane = tid & 31;
  const int lrow = lane & 15;        // A row / B column within tile
  const int hi   = lane >> 4;        // half-wave id
  const int k8   = hi * 8;           // A-fragment K offset
  const int k16  = hi * 16;          // B-fragment K offset
  const int dmb  = hi * 8;           // D-fragment M base: m = vgpr + dmb

  char* wb = smem + WAVE_BYTES_OFF + (size_t)wave * WAVE_BYTES;
  _Float16* sx1v = reinterpret_cast<_Float16*>(wb);  // [3][16][64]
  _Float16* sdv  = sx1v + 3 * 16 * 64;               // [16][64]
  _Float16* sh   = sdv + 16 * 64;                    // [16][64]
  float*    sx2  = reinterpret_cast<float*>(wb + 5120 * 2); // [16][4] = {x2s, x2v0..2}

  const long nTiles = (Etot + 15) >> 4;
  for (long tile = (long)blockIdx.x * NW + wave; tile < nTiles; tile += (long)gridDim.x * NW) {
    const long e0 = tile << 4;
    const long rowm = e0 + lrow;
    const bool rvalid = rowm < Etot;
    const long rclamp = rvalid ? rowm : (Etot - 1);

    // ---- stage x2 (per-row scalars) ----
    {
      v4f f2 = *reinterpret_cast<const v4f*>(fea_in2 + rclamp * 4);
#pragma unroll
      for (int i = 0; i < 4; ++i) sx2[lrow * 4 + i] = rvalid ? f2[i] : 0.f;
    }

    // ---- stage x1v (register de-interleave) + dv, two lanes per row ----
    {
      const int r  = lane >> 1;          // row 0..15
      const int ub = (lane & 1) * 32;    // u base 0 or 32 (96 interleaved floats per lane)
      const long er = e0 + r;
      const bool ok = er < Etot;
      const long erc = ok ? er : (Etot - 1);
      const float* src = fea_in1 + erc * ROW_IN1 + MUL_S + ub * 3;
      const float xv0 = sx2[r * 4 + 1];
      const float xv1 = sx2[r * 4 + 2];
      const float xv2 = sx2[r * 4 + 3];
#pragma unroll
      for (int c4 = 0; c4 < 4; ++c4) {   // 4 chunks of 24 floats = 8 u's each
        float fv[24];
#pragma unroll
        for (int t = 0; t < 6; ++t) {
          v4f q = *reinterpret_cast<const v4f*>(src + c4 * 24 + t * 4);
#pragma unroll
          for (int e = 0; e < 4; ++e) fv[4 * t + e] = q[e];
        }
        if (!ok) {
#pragma unroll
          for (int t = 0; t < 24; ++t) fv[t] = 0.f;
        }
        union Pk { _Float16 h[8]; uint4 q128; };
        Pk p0, p1, p2c, pd;
#pragma unroll
        for (int t = 0; t < 8; ++t) {
          p0.h[t]  = (_Float16)fv[3 * t + 0];
          p1.h[t]  = (_Float16)fv[3 * t + 1];
          p2c.h[t] = (_Float16)fv[3 * t + 2];
          pd.h[t]  = (_Float16)(fv[3 * t] * xv0 + fv[3 * t + 1] * xv1 + fv[3 * t + 2] * xv2);
        }
        const int uoff = ub + c4 * 8;
        *reinterpret_cast<uint4*>(sx1v + (0 * 16 + r) * 64 + uoff) = p0.q128;
        *reinterpret_cast<uint4*>(sx1v + (1 * 16 + r) * 64 + uoff) = p1.q128;
        *reinterpret_cast<uint4*>(sx1v + (2 * 16 + r) * 64 + uoff) = p2c.q128;
        *reinterpret_cast<uint4*>(sdv + r * 64 + uoff)             = pd.q128;
      }
    }

    // per-lane row scalars in D layout (m = v + dmb)
    float d_x2s[8], d_x2v0[8], d_x2v1[8], d_x2v2[8];
#pragma unroll
    for (int v = 0; v < 8; ++v) {
      int m = v + dmb;
      d_x2s[v]  = sx2[m * 4 + 0];
      d_x2v0[v] = sx2[m * 4 + 1];
      d_x2v1[v] = sx2[m * 4 + 2];
      d_x2v2[v] = sx2[m * 4 + 3];
    }

    // ================= FC chain: 128 -> 64 -> 64 -> (192 lazily) =================
    v16h aF[4];
#pragma unroll
    for (int ks = 0; ks < 4; ++ks) aF[ks] = fragA_gbl(fea_weight, rclamp, FC_IN, ks * 32 + k8, rvalid);

#pragma unroll
    for (int j = 0; j < 4; ++j) {           // H0 = silu(FW @ fc_w0 + b0)
      v8f acc = zf();
#pragma unroll
      for (int ks = 0; ks < 4; ++ks)
        acc = wmma_f16(aF[ks], fragB_lds(wF0, 128, j * 16 + lrow, ks * 32 + k16), acc);
      int n = j * 16 + lrow;
      float bias = sb0[n];
#pragma unroll
      for (int v = 0; v < 8; ++v) {
        float x = acc[v] + bias;
        sh[(v + dmb) * 64 + n] = (_Float16)(x * sigmoidf_(x));
      }
    }
    v16h aH[2];
#pragma unroll
    for (int ks = 0; ks < 2; ++ks) aH[ks] = fragA_lds(sh, 64, lrow, ks * 32 + k8);

#pragma unroll
    for (int j = 0; j < 4; ++j) {           // H1 = silu(H0 @ fc_w1 + b1)
      v8f acc = zf();
#pragma unroll
      for (int ks = 0; ks < 2; ++ks)
        acc = wmma_f16(aH[ks], fragB_lds(wF1, 64, j * 16 + lrow, ks * 32 + k16), acc);
      int n = j * 16 + lrow;
      float bias = sb1[n];
#pragma unroll
      for (int v = 0; v < 8; ++v) {
        float x = acc[v] + bias;
        sh[(v + dmb) * 64 + n] = (_Float16)(x * sigmoidf_(x));
      }
    }
    v16h aH1[2];
#pragma unroll
    for (int ks = 0; ks < 2; ++ks) aH1[ks] = fragA_lds(sh, 64, lrow, ks * 32 + k8);

    // ================= scalar outputs: cols 0..127 =================
    v16h aS[4];
#pragma unroll
    for (int ks = 0; ks < 4; ++ks) aS[ks] = fragA_gbl(fea_in1, rclamp, ROW_IN1, ks * 32 + k8, rvalid);
    v16h aD[2];
#pragma unroll
    for (int ks = 0; ks < 2; ++ks) aD[ks] = fragA_lds(sdv, 64, lrow, ks * 32 + k8);

#pragma unroll
    for (int j = 0; j < 8; ++j) {
      v8f s0 = zf(), s4 = zf(), wj = zf();
#pragma unroll
      for (int ks = 0; ks < 4; ++ks)
        s0 = wmma_f16(aS[ks], fragB_lds(wT0, 128, j * 16 + lrow, ks * 32 + k16), s0);
#pragma unroll
      for (int ks = 0; ks < 2; ++ks) {
        s4 = wmma_f16(aD[ks],  fragB_lds(wT4, 64, j * 16 + lrow, ks * 32 + k16), s4);
        wj = wmma_f16(aH1[ks], fragB_lds(wF2, 64, j * 16 + lrow, ks * 32 + k16), wj);
      }
      int n = j * 16 + lrow;
      float c0 = sw2_0[n] * INV_S;
      float c4 = sw2_4[n] * (INV_V * SQ3);
      float bw = sb2[n];
#pragma unroll
      for (int v = 0; v < 8; ++v) {
        float scal = (s0[v] * d_x2s[v] * c0 + s4[v] * c4) * SQ2;
        float o = scal * sigmoidf_(scal) * (wj[v] + bw);
        long er = e0 + v + dmb;
        if (er < Etot) out[er * ROW_OUT + n] = o;
      }
    }

    // ================= gate/vector outputs: cols 128..319 =================
    v16h aV0[2], aV1[2], aV2[2];
#pragma unroll
    for (int ks = 0; ks < 2; ++ks) {
      aV0[ks] = fragA_lds(sx1v + 0 * 16 * 64, 64, lrow, ks * 32 + k8);
      aV1[ks] = fragA_lds(sx1v + 1 * 16 * 64, 64, lrow, ks * 32 + k8);
      aV2[ks] = fragA_lds(sx1v + 2 * 16 * 64, 64, lrow, ks * 32 + k8);
    }
#pragma unroll
    for (int j = 0; j < 4; ++j) {
      v8f g1 = zf(), g5 = zf(), p2 = zf(), p30 = zf(), p31 = zf(), p32 = zf(), wj = zf();
#pragma unroll
      for (int ks = 0; ks < 4; ++ks) {
        g1 = wmma_f16(aS[ks], fragB_lds(wT1, 128, j * 16 + lrow, ks * 32 + k16), g1);
        p2 = wmma_f16(aS[ks], fragB_lds(wT2, 128, j * 16 + lrow, ks * 32 + k16), p2);
      }
#pragma unroll
      for (int ks = 0; ks < 2; ++ks) {
        g5  = wmma_f16(aD[ks],  fragB_lds(wT5, 64, j * 16 + lrow, ks * 32 + k16), g5);
        p30 = wmma_f16(aV0[ks], fragB_lds(wT3, 64, j * 16 + lrow, ks * 32 + k16), p30);
        p31 = wmma_f16(aV1[ks], fragB_lds(wT3, 64, j * 16 + lrow, ks * 32 + k16), p31);
        p32 = wmma_f16(aV2[ks], fragB_lds(wT3, 64, j * 16 + lrow, ks * 32 + k16), p32);
        wj  = wmma_f16(aH1[ks], fragB_lds(wF2, 64, (8 + j) * 16 + lrow, ks * 32 + k16), wj);
      }
      int u = j * 16 + lrow;
      float c1 = sw2_1[u] * INV_S;
      float c5 = sw2_5[u] * (INV_V * SQ3);
      float c2 = sw2_2[u] * INV_S;
      float c3 = sw2_3[u] * INV_V;
      float bw = sb2[MUL_S + u];
#pragma unroll
      for (int v = 0; v < 8; ++v) {
        float gate = sigmoidf_((g1[v] * d_x2s[v] * c1 + g5[v] * c5) * SQ2);
        float fm = gate * (wj[v] + bw);
        long er = e0 + v + dmb;
        if (er < Etot) {
          float* op = out + er * ROW_OUT + MUL_S + 3 * u;
          op[0] = fm * ((p2[v] * d_x2v0[v] * c2 + p30[v] * d_x2s[v] * c3) * SQ2);
          op[1] = fm * ((p2[v] * d_x2v1[v] * c2 + p31[v] * d_x2s[v] * c3) * SQ2);
          op[2] = fm * ((p2[v] * d_x2v2[v] * c2 + p32[v] * d_x2s[v] * c3) * SQ2);
        }
      }
    }
  }
}

extern "C" void kernel_launch(void* const* d_in, const int* in_sizes, int n_in,
                              void* d_out, int out_size, void* d_ws, size_t ws_size,
                              hipStream_t stream) {
  (void)n_in; (void)out_size;
  const float* fea_in1    = (const float*)d_in[0];
  const float* fea_in2    = (const float*)d_in[1];
  const float* fea_weight = (const float*)d_in[2];
  // d_in[3] = batch_edge (unused by the reference)
  const float* w1_p0 = (const float*)d_in[4];
  const float* w2_p0 = (const float*)d_in[5];
  const float* w1_p1 = (const float*)d_in[6];
  const float* w2_p1 = (const float*)d_in[7];
  const float* w1_p2 = (const float*)d_in[8];
  const float* w2_p2 = (const float*)d_in[9];
  const float* w1_p3 = (const float*)d_in[10];
  const float* w2_p3 = (const float*)d_in[11];
  const float* w1_p4 = (const float*)d_in[12];
  const float* w2_p4 = (const float*)d_in[13];
  const float* w1_p5 = (const float*)d_in[14];
  const float* w2_p5 = (const float*)d_in[15];
  const float* fc_w0 = (const float*)d_in[16];
  const float* fc_b0 = (const float*)d_in[17];
  const float* fc_w1 = (const float*)d_in[18];
  const float* fc_b1 = (const float*)d_in[19];
  const float* fc_w2 = (const float*)d_in[20];
  const float* fc_b2 = (const float*)d_in[21];
  float* out = (float*)d_out;

  const long E = (long)in_sizes[3];   // batch_edge has shape (E,)
  const long nTiles = (E + 15) / 16;
  long blocksNeeded = (nTiles + NW - 1) / NW;
  int blocks = (int)(blocksNeeded < 256 ? blocksNeeded : 256);  // grid-stride amortizes preload

  const int mode = (d_ws != nullptr && ws_size >= (size_t)BLOB_BYTES) ? 1 : 0;
  if (mode == 1) {
    build_blob_kernel<<<64, 256, 0, stream>>>(
        w1_p0, w2_p0, w1_p1, w2_p1, w1_p2, w2_p2, w1_p3, w2_p3,
        w1_p4, w2_p4, w1_p5, w2_p5, fc_w0, fc_b0, fc_w1, fc_b1, fc_w2, fc_b2,
        (char*)d_ws);
  }

  // allow >64KB dynamic LDS (gfx1250 WGP supports 320KB workgroup LDS)
  (void)hipFuncSetAttribute((const void*)equiconv_wmma_kernel,
                            hipFuncAttributeMaxDynamicSharedMemorySize, (int)SMEM_BYTES);

  equiconv_wmma_kernel<<<blocks, NTHR, SMEM_BYTES, stream>>>(
      fea_in1, fea_in2, fea_weight,
      w1_p0, w2_p0, w1_p1, w2_p1, w1_p2, w2_p2,
      w1_p3, w2_p3, w1_p4, w2_p4, w1_p5, w2_p5,
      fc_w0, fc_b0, fc_w1, fc_b1, fc_w2, fc_b2,
      (const char*)d_ws, mode, out, E);
}
